// MPAMMambaFusion_9749575762051
// MI455X (gfx1250) — matrix operand order repaired and checked by reading, not compiled, not run on hardware
//
#include <hip/hip_runtime.h>
#include <hip/hip_bf16.h>

// ---------------------------------------------------------------------------
// MPAM + Mamba2D fusion for gfx1250 (MI455X).
// bf16 WMMA (v_wmma_f32_16x16x32_bf16) for all 5 GEMMs, fp32 accumulate.
// Position-major (B, L, C) layout so every GEMM is NT with contiguous K.
// Scan: 1 thread per (b,d,s), d_state reduction via __shfl_xor in half-wave.
// GEMM grid is exact (no tail), tile count per wave is a template constant:
// no divergent EXEC around WMMA (ISA requires EXEC all-ones), no spills.
// ---------------------------------------------------------------------------

#define B_    2
#define CR_   256
#define CS_   512
#define HR_   64
#define WR_   64
#define HS_   32
#define WS_   32
#define L_    (HR_ * WR_)      // 4096
#define DI_   512
#define DS_   16
#define DTR_  16
#define BL_   (B_ * L_)        // 8192
#define EPS_  1e-5f

typedef __attribute__((ext_vector_type(16))) __bf16 v16bf;
typedef __attribute__((ext_vector_type(8)))  __bf16 v8bf;
typedef __attribute__((ext_vector_type(8)))  float  v8f;

// ---------------------------------------------------------------------------
// NT bf16 GEMM:  C[M,N] = A[M,K] * B[N,K]^T,  fp32 accumulate.
// A row-major (lda=K), B row-major (ldb=K), C row-major (ldc=N).
// Requires: M%16==0, K%32==0, N == ntc * NTILES * 16 for integer ntc,
// and (M/16)*ntc divisible by 8 (exact grid, launched by host helper).
// One wave computes a 16(M) x (16*NTILES)(N) strip; A-frag reused NTILES x.
// A/B frags use contiguous k (lane half h: k=16h..16h+15); f32 C 16x16 per ISA 7.12.2.
// ---------------------------------------------------------------------------
template <int NTILES>
__global__ __launch_bounds__(256) void k_gemm_nt_bf16(
        const __bf16* __restrict__ A,
        const __bf16* __restrict__ Bm,
        float* __restrict__ C,
        int M, int N, int K) {
    const int lane = threadIdx.x & 31;
    const int wave = threadIdx.x >> 5;

    const int ntc  = N / (16 * NTILES);     // col groups per row
    const int tile = blockIdx.x * 8 + wave; // wave-uniform, always in range
    const int rt   = tile / ntc;
    const int cg   = tile - rt * ntc;
    const int row0 = rt << 4;
    const int col0 = cg * (16 * NTILES);

    v8f acc[NTILES];
#pragma unroll
    for (int j = 0; j < NTILES; ++j) acc[j] = v8f{};

    const int n16  = lane & 15;
    const int aoff = (lane < 16) ? 0 : 16;  // A: lanes 0-15 K=0..15; lanes 16-31 K=16..31
    const int boff = (lane < 16) ? 0 : 16;  // B: lanes 0-15 K=0..15; lanes 16-31 K=16..31

    const __bf16* arow = A + (size_t)(row0 + n16) * K + aoff;
    const __bf16* bcol = Bm + (size_t)(col0 + n16) * K + boff;

    for (int kb = 0; kb < K; kb += 32) {
        v16bf afrag;
        ((v8bf*)&afrag)[0] = *(const v8bf*)(arow + kb);
        ((v8bf*)&afrag)[1] = *(const v8bf*)(arow + kb + 8);
#pragma unroll
        for (int j = 0; j < NTILES; ++j) {
            v16bf bfrag = *(const v16bf*)(bcol + (size_t)(j * 16) * K + kb);
            acc[j] = __builtin_amdgcn_wmma_f32_16x16x32_bf16(
                false, afrag, false, bfrag, (short)0, acc[j], false, false);
        }
    }

    const int mbase = (lane < 16) ? 0 : 8;  // f32 C/D: vgpr r -> M = r + (hi?8:0)
#pragma unroll
    for (int j = 0; j < NTILES; ++j) {
        float* ccol = C + (size_t)(row0 + mbase) * N + col0 + j * 16 + n16;
#pragma unroll
        for (int r = 0; r < 8; ++r)
            ccol[(size_t)r * N] = acc[j][r];
    }
}

// --------------------------- small prep kernels ----------------------------

__global__ void k_f32_to_bf16(const float* __restrict__ in,
                              __bf16* __restrict__ out, int n) {
    int i = blockIdx.x * blockDim.x + threadIdx.x;
    if (i < n) out[i] = (__bf16)in[i];
}

// dt_proj_w (512x16) -> zero-padded bf16 (512x32) so K=32 for WMMA
__global__ void k_dtw_pad(const float* __restrict__ w, __bf16* __restrict__ out) {
    int i = blockIdx.x * blockDim.x + threadIdx.x; // 512*32
    int col = i & 31, r = i >> 5;
    out[i] = (__bf16)((col < DTR_) ? w[r * DTR_ + col] : 0.f);
}

// scp_feat (b,512,32,32) -> scpT bf16 (b, 1024, 512)  (position-major)
__global__ void k_transpose_scp(const float* __restrict__ scp,
                                __bf16* __restrict__ out) {
    int i = blockIdx.x * blockDim.x + threadIdx.x;   // 2*1024*512 = 2^20
    int c = i & 511;
    int p = (i >> 9) & 1023;
    int b = i >> 19;
    out[i] = (__bf16)scp[((size_t)b * CS_ + c) * (HS_ * WS_) + p];
}

// fused(b,l,c) = BN(resnet) + bilinear_upsample(BN(spT))   (BN commutes w/ lerp)
__global__ void k_fuse(const float* __restrict__ resnet,
                       const float* __restrict__ spT,
                       const float* __restrict__ rg, const float* __restrict__ rb,
                       const float* __restrict__ rm, const float* __restrict__ rv,
                       const float* __restrict__ sg, const float* __restrict__ sb,
                       const float* __restrict__ sm, const float* __restrict__ sv,
                       float* __restrict__ fused) {
    int tid = blockIdx.x * blockDim.x + threadIdx.x; // 2*4096*256 = 2^21
    int c = tid & 255;
    int l = (tid >> 8) & (L_ - 1);
    int b = tid >> 20;
    int y = l >> 6, x = l & 63;

    float rinv = rg[c] * rsqrtf(rv[c] + EPS_);
    float rn = resnet[(((size_t)b * CR_ + c) * HR_ + y) * WR_ + x] * rinv
               + (rb[c] - rm[c] * rinv);

    // half-pixel bilinear, 32 -> 64
    float sy = (y + 0.5f) * 0.5f - 0.5f;
    float sx = (x + 0.5f) * 0.5f - 0.5f;
    int y0 = (int)floorf(sy); float fy = sy - (float)y0;
    int x0 = (int)floorf(sx); float fx = sx - (float)x0;
    int y0c = max(0, min(HS_ - 1, y0)), y1c = max(0, min(HS_ - 1, y0 + 1));
    int x0c = max(0, min(WS_ - 1, x0)), x1c = max(0, min(WS_ - 1, x0 + 1));
    const float* spb = spT + (size_t)b * (HS_ * WS_) * CR_;
    float v00 = spb[(y0c * WS_ + x0c) * CR_ + c];
    float v01 = spb[(y0c * WS_ + x1c) * CR_ + c];
    float v10 = spb[(y1c * WS_ + x0c) * CR_ + c];
    float v11 = spb[(y1c * WS_ + x1c) * CR_ + c];
    float vi = (1.f - fy) * ((1.f - fx) * v00 + fx * v01)
             + fy * ((1.f - fx) * v10 + fx * v11);

    float sinv = sg[c] * rsqrtf(sv[c] + EPS_);
    fused[tid] = rn + vi * sinv + (sb[c] - sm[c] * sinv);
}

// LayerNorm over 256 channels, one wave per row, output bf16 for GEMM A.
__global__ void k_layernorm(const float* __restrict__ x,
                            const float* __restrict__ g, const float* __restrict__ bt,
                            __bf16* __restrict__ out, int rows) {
    int wave = threadIdx.x >> 5, lane = threadIdx.x & 31;
    int row = blockIdx.x * 8 + wave;
    if (row >= rows) return;
    const float* xr = x + (size_t)row * CR_;
    float v[8], s = 0.f, s2 = 0.f;
#pragma unroll
    for (int i = 0; i < 8; ++i) {
        v[i] = xr[lane * 8 + i];
        s += v[i]; s2 += v[i] * v[i];
    }
#pragma unroll
    for (int m = 16; m > 0; m >>= 1) {
        s  += __shfl_xor(s,  m, 32);
        s2 += __shfl_xor(s2, m, 32);
    }
    float mean = s * (1.f / CR_);
    float rstd = rsqrtf(s2 * (1.f / CR_) - mean * mean + EPS_);
    __bf16* orow = out + (size_t)row * CR_;
#pragma unroll
    for (int i = 0; i < 8; ++i) {
        int c = lane * 8 + i;
        orow[c] = (__bf16)((v[i] - mean) * rstd * g[c] + bt[c]);
    }
}

// causal depthwise conv(width 3) over L + SiLU; writes u (f32) and u (bf16)
__global__ void k_conv_silu(const float* __restrict__ xz,
                            const float* __restrict__ cw, const float* __restrict__ cb,
                            float* __restrict__ u32, __bf16* __restrict__ ubf) {
    int tid = blockIdx.x * blockDim.x + threadIdx.x; // 2*4096*512 = 2^22
    int d = tid & 511;
    int l = (tid >> 9) & (L_ - 1);
    int b = tid >> 21;
    const float* col = xz + (size_t)(b * L_) * (2 * DI_) + d; // xm = first DI_ cols
    float acc = cb[d];
    float w0 = cw[d * 3 + 0], w1 = cw[d * 3 + 1], w2 = cw[d * 3 + 2];
    if (l >= 2) acc += w0 * col[(size_t)(l - 2) * (2 * DI_)];
    if (l >= 1) acc += w1 * col[(size_t)(l - 1) * (2 * DI_)];
    acc += w2 * col[(size_t)l * (2 * DI_)];
    float uu = acc / (1.f + __expf(-acc));
    u32[tid] = uu;
    ubf[tid] = (__bf16)uu;
}

// x_dbl cols 0..15 -> zero-padded bf16 (BL x 32) for dt_proj GEMM
__global__ void k_dt_prep(const float* __restrict__ xdbl, __bf16* __restrict__ out) {
    int i = blockIdx.x * blockDim.x + threadIdx.x; // 8192*32
    int col = i & 31, row = i >> 5;
    out[i] = (__bf16)((col < DTR_) ? xdbl[(size_t)row * 48 + col] : 0.f);
}

// Selective scan. thread = (b, d, s); s-lanes share a half-wave.
// h[l] = exp(dt*A)*h[l-1] + dt*B*u ; y[l,d] = sum_s h*C + u*D
__global__ void k_scan(const float* __restrict__ dt_raw, const float* __restrict__ dt_b,
                       const float* __restrict__ u, const float* __restrict__ xdbl,
                       const float* __restrict__ A_log, const float* __restrict__ Dp,
                       float* __restrict__ ypart) {
    int tid = blockIdx.x * blockDim.x + threadIdx.x; // 2*512*16 = 16384
    int s = tid & 15;
    int G = tid >> 4;
    int b = G >> 9;
    int d = G & 511;

    float Aval = -__expf(A_log[d * DS_ + s]);
    float bias = dt_b[d];
    float Dv = Dp[d];

    const float* dtp = dt_raw + (size_t)(b * L_) * DI_ + d;
    const float* up  = u      + (size_t)(b * L_) * DI_ + d;
    const float* xdp = xdbl   + (size_t)(b * L_) * 48;
    float*       yp  = ypart  + (size_t)(b * L_) * DI_ + d;

    float h = 0.f;
    for (int l = 0; l < L_; ++l) {
        float dtr = dtp[(size_t)l * DI_] + bias;
        float dt = (dtr > 20.f) ? dtr : log1pf(__expf(dtr)); // softplus (idempotent recompute)
        float uu = up[(size_t)l * DI_];
        float Bc = xdp[(size_t)l * 48 + DTR_ + s];
        float Cc = xdp[(size_t)l * 48 + DTR_ + DS_ + s];
        float dA = __expf(dt * Aval);
        h = fmaf(dA, h, dt * Bc * uu);
        float p = h * Cc;
        p += __shfl_xor(p, 1, 32);
        p += __shfl_xor(p, 2, 32);
        p += __shfl_xor(p, 4, 32);
        p += __shfl_xor(p, 8, 32);
        if (s == 0) yp[(size_t)l * DI_] = fmaf(uu, Dv, p);
    }
}

// y = y_partial * silu(z), bf16 for out_proj GEMM
__global__ void k_gate(const float* __restrict__ ypart, const float* __restrict__ xz,
                       __bf16* __restrict__ ybf) {
    int tid = blockIdx.x * blockDim.x + threadIdx.x; // 2^22
    int d = tid & 511;
    int l = (tid >> 9) & (L_ - 1);
    int b = tid >> 21;
    float z = xz[((size_t)(b * L_ + l)) * (2 * DI_) + DI_ + d];
    float sz = z / (1.f + __expf(-z));
    ybf[tid] = (__bf16)(ypart[tid] * sz);
}

// d_out (NCHW) = fused + out_seq  (both stored position-major)
__global__ void k_final(const float* __restrict__ fused, const float* __restrict__ oseq,
                        float* __restrict__ out) {
    int tid = blockIdx.x * blockDim.x + threadIdx.x; // 2^21
    int x = tid & 63;
    int y = (tid >> 6) & 63;
    int c = (tid >> 12) & 255;
    int b = tid >> 20;
    size_t idx = ((size_t)(b * L_) + (size_t)(y * WR_ + x)) * CR_ + c;
    out[tid] = fused[idx] + oseq[idx];
}

// ---------------------------------------------------------------------------

static inline void launch_gemm(const __bf16* A, const __bf16* Bm, float* C,
                               int M, int N, int K, hipStream_t s) {
    if (N % 64 == 0) {
        int waves = (M >> 4) * (N >> 6);
        k_gemm_nt_bf16<4><<<waves / 8, 256, 0, s>>>(A, Bm, C, M, N, K);
    } else {
        // N multiple of 48 (x_proj): 3 tiles per wave
        int waves = (M >> 4) * (N / 48);
        k_gemm_nt_bf16<3><<<waves / 8, 256, 0, s>>>(A, Bm, C, M, N, K);
    }
}

extern "C" void kernel_launch(void* const* d_in, const int* in_sizes, int n_in,
                              void* d_out, int out_size, void* d_ws, size_t ws_size,
                              hipStream_t stream) {
    const float* resnet   = (const float*)d_in[0];
    const float* scp      = (const float*)d_in[1];
    const float* scp_w    = (const float*)d_in[2];
    const float* scp_bn_g = (const float*)d_in[3];
    const float* scp_bn_b = (const float*)d_in[4];
    const float* scp_bn_m = (const float*)d_in[5];
    const float* scp_bn_v = (const float*)d_in[6];
    const float* res_bn_g = (const float*)d_in[7];
    const float* res_bn_b = (const float*)d_in[8];
    const float* res_bn_m = (const float*)d_in[9];
    const float* res_bn_v = (const float*)d_in[10];
    const float* ln_g     = (const float*)d_in[11];
    const float* ln_b     = (const float*)d_in[12];
    const float* in_w     = (const float*)d_in[13];
    const float* conv_w   = (const float*)d_in[14];
    const float* conv_b   = (const float*)d_in[15];
    const float* xp_w     = (const float*)d_in[16];
    const float* dt_w     = (const float*)d_in[17];
    const float* dt_bias  = (const float*)d_in[18];
    const float* A_log    = (const float*)d_in[19];
    const float* Dvec     = (const float*)d_in[20];
    const float* out_w    = (const float*)d_in[21];

    char* ws = (char*)d_ws;
    size_t off = 0;
    auto take = [&](size_t bytes) -> char* {
        char* p = ws + off;
        off += (bytes + 255) & ~(size_t)255;
        return p;
    };

    __bf16* scp_w_bf = (__bf16*)take((size_t)CR_ * CS_ * 2);
    __bf16* in_w_bf  = (__bf16*)take((size_t)2 * DI_ * CR_ * 2);
    __bf16* xp_w_bf  = (__bf16*)take((size_t)48 * DI_ * 2);
    __bf16* dt_w_bf  = (__bf16*)take((size_t)DI_ * 32 * 2);
    __bf16* out_w_bf = (__bf16*)take((size_t)CR_ * DI_ * 2);
    __bf16* scpT     = (__bf16*)take((size_t)B_ * 1024 * CS_ * 2);
    float*  spT      = (float*) take((size_t)B_ * 1024 * CR_ * 4);
    float*  fused    = (float*) take((size_t)BL_ * CR_ * 4);
    __bf16* xn_bf    = (__bf16*)take((size_t)BL_ * CR_ * 2);
    float*  xz       = (float*) take((size_t)BL_ * 2 * DI_ * 4);
    float*  u_f32    = (float*) take((size_t)BL_ * DI_ * 4);
    __bf16* u_bf     = (__bf16*)take((size_t)BL_ * DI_ * 2);
    float*  x_dbl    = (float*) take((size_t)BL_ * 48 * 4);
    __bf16* dtpad    = (__bf16*)take((size_t)BL_ * 32 * 2);
    float*  dt_raw   = (float*) take((size_t)BL_ * DI_ * 4);
    float*  y_part   = (float*) take((size_t)BL_ * DI_ * 4);
    __bf16* y_bf     = (__bf16*)take((size_t)BL_ * DI_ * 2);
    float*  out_seq  = (float*) take((size_t)BL_ * CR_ * 4);

    // weight conversions
    k_f32_to_bf16<<<(CR_*CS_ + 255)/256, 256, 0, stream>>>(scp_w, scp_w_bf, CR_*CS_);
    k_f32_to_bf16<<<(2*DI_*CR_ + 255)/256, 256, 0, stream>>>(in_w, in_w_bf, 2*DI_*CR_);
    k_f32_to_bf16<<<(48*DI_ + 255)/256, 256, 0, stream>>>(xp_w, xp_w_bf, 48*DI_);
    k_f32_to_bf16<<<(CR_*DI_ + 255)/256, 256, 0, stream>>>(out_w, out_w_bf, CR_*DI_);
    k_dtw_pad<<<(DI_*32 + 255)/256, 256, 0, stream>>>(dt_w, dt_w_bf);

    // scp 1x1 conv as NT GEMM in position-major layout
    k_transpose_scp<<<(B_*1024*CS_)/256, 256, 0, stream>>>(scp, scpT);
    launch_gemm(scpT, scp_w_bf, spT, B_ * 1024, CR_, CS_, stream);

    // BN + bilinear + residual fuse -> fused (B, L, CR)
    k_fuse<<<(BL_*CR_)/256, 256, 0, stream>>>(resnet, spT,
        res_bn_g, res_bn_b, res_bn_m, res_bn_v,
        scp_bn_g, scp_bn_b, scp_bn_m, scp_bn_v, fused);

    // LayerNorm -> bf16 ; in_proj GEMM
    k_layernorm<<<(BL_ + 7)/8, 256, 0, stream>>>(fused, ln_g, ln_b, xn_bf, BL_);
    launch_gemm(xn_bf, in_w_bf, xz, BL_, 2 * DI_, CR_, stream);

    // depthwise causal conv + silu
    k_conv_silu<<<(BL_*DI_)/256, 256, 0, stream>>>(xz, conv_w, conv_b, u_f32, u_bf);

    // x_proj GEMM (N=48, 3 tiles/wave) ; dt_proj GEMM (K padded to 32)
    launch_gemm(u_bf, xp_w_bf, x_dbl, BL_, 48, DI_, stream);
    k_dt_prep<<<(BL_*32)/256, 256, 0, stream>>>(x_dbl, dtpad);
    launch_gemm(dtpad, dt_w_bf, dt_raw, BL_, DI_, 32, stream);

    // selective scan
    k_scan<<<(B_*DI_*DS_)/256, 256, 0, stream>>>(dt_raw, dt_bias, u_f32, x_dbl,
                                                 A_log, Dvec, y_part);

    // gate + out_proj GEMM + final residual
    k_gate<<<(BL_*DI_)/256, 256, 0, stream>>>(y_part, xz, y_bf);
    launch_gemm(y_bf, out_w_bf, out_seq, BL_, CR_, DI_, stream);
    k_final<<<(BL_*CR_)/256, 256, 0, stream>>>(fused, out_seq, (float*)d_out);
}
